// SipMaskPostProcessor_51874615001702
// MI455X (gfx1250) — compile-verified, matrix operand order checked by reading, not run on hardware
//
#include <hip/hip_runtime.h>
#include <math.h>

// ---------------------------------------------------------------------------
// SipMask post-processor for MI455X (gfx1250, wave32, WMMA)
// Mask-head einsum -> v_wmma_f32_16x16x32_f16 (M=16 pixels, N=16 dets, K=32).
// A/B fragments fed by contiguous b128 loads of pre-converted f16 data.
// ---------------------------------------------------------------------------

typedef __attribute__((ext_vector_type(16))) _Float16 v16h;
typedef __attribute__((ext_vector_type(8)))  _Float16 v8h;
typedef __attribute__((ext_vector_type(8)))  float    v8f;

#define NIMG      2
#define NCLS      80
#define HF        60
#define WF        80
#define NPOS      (HF * WF)        // 4800
#define NSCORE    (NPOS * NCLS)    // 384000
#define IMGH      480
#define IMGW      640
#define HM_       240
#define WM_       320
#define NPIX      (HM_ * WM_)      // 76800
#define PRE_TOPK  1000
#define POST_TOPK 100
#define DPAD      112              // detections padded to 7 tiles of 16
#define NMS_T     0.6f
#define MASK_T    0.4f
#define SCALE_    2.0f
#define CLS_OFF   4096.0f

// mask GEMM tiling: wave = 16x16 tile, 4 M-tiles per wave (B reuse)
#define NTILES_N  7
#define NTILES_M  (NPIX / 16)              // 4800
#define MT_PER_WAVE 4
#define WAVES_TOT (NTILES_N * NTILES_M / MT_PER_WAVE)   // 8400
#define MASK_BLOCKS (WAVES_TOT / 8)                     // 1050

// ---- workspace layout (floats, then ints, then halfs) ----
#define WSF_FLAT  0                              // NIMG*NSCORE
#define WSF_TOPV  (WSF_FLAT + NIMG * NSCORE)     // NIMG*PRE_TOPK
#define WSF_BOX   (WSF_TOPV + NIMG * PRE_TOPK)   // NIMG*PRE_TOPK*4
#define WSF_BOFF  (WSF_BOX  + NIMG * PRE_TOPK * 4)
#define WSF_KSCR  (WSF_BOFF + NIMG * PRE_TOPK * 4)
#define WSF_V2    (WSF_KSCR + NIMG * PRE_TOPK)   // NIMG*POST_TOPK
#define WSF_FVAL  (WSF_V2   + NIMG * POST_TOPK)
#define WSF_FB    (WSF_FVAL + NIMG * POST_TOPK)  // NIMG*DPAD*4 (zero padded)
#define WSF_END   (WSF_FB   + NIMG * DPAD * 4)

#define WSI_TOPI  0                              // NIMG*PRE_TOPK
#define WSI_KEEP  (WSI_TOPI + NIMG * PRE_TOPK)
#define WSI_SEL   (WSI_KEEP + NIMG * PRE_TOPK)   // NIMG*POST_TOPK
#define WSI_END   (WSI_SEL  + NIMG * POST_TOPK)

#define WSH_COEF  0                                      // NIMG*DPAD*128
#define WSH_PM    (WSH_COEF + NIMG * DPAD * 128)         // NIMG*POST_TOPK*NPIX
#define WSH_FEAT  (WSH_PM   + NIMG * POST_TOPK * NPIX)   // NIMG*NPIX*32

__device__ __forceinline__ float sigmoidf_(float x) {
    return 1.0f / (1.0f + expf(-x));
}

// ---------------------------------------------------------------------------
// 0) transpose+convert feat (NIMG,32,HM,WM) f32 -> featT (NIMG,NPIX,32) f16
// ---------------------------------------------------------------------------
__global__ void k_feat_f16(const float* __restrict__ feat,
                           _Float16* __restrict__ featT) {
    int gid = blockIdx.x * blockDim.x + threadIdx.x;   // img*NPIX + p
    if (gid >= NIMG * NPIX) return;
    int img = gid / NPIX;
    int p   = gid % NPIX;
    const float* fb = feat + img * 32 * NPIX + p;
    v8h v0, v1, v2, v3;
#pragma unroll
    for (int c = 0; c < 8; ++c) v0[c] = (_Float16)fb[c * NPIX];
#pragma unroll
    for (int c = 0; c < 8; ++c) v1[c] = (_Float16)fb[(c + 8) * NPIX];
#pragma unroll
    for (int c = 0; c < 8; ++c) v2[c] = (_Float16)fb[(c + 16) * NPIX];
#pragma unroll
    for (int c = 0; c < 8; ++c) v3[c] = (_Float16)fb[(c + 24) * NPIX];
    v8h* dst = (v8h*)(featT + (size_t)gid * 32);
    dst[0] = v0; dst[1] = v1; dst[2] = v2; dst[3] = v3;
}

// ---------------------------------------------------------------------------
// 1) flat candidate scores: flat[img][pos*C + c] = cand ? sig(cls)*sig(ctr) : -1
// ---------------------------------------------------------------------------
__global__ void k_score_flat(const float* __restrict__ box_cls,
                             const float* __restrict__ centerness,
                             float* __restrict__ flat) {
    int gid = blockIdx.x * blockDim.x + threadIdx.x;
    if (gid >= NIMG * NSCORE) return;
    int img = gid / NSCORE;
    int r   = gid % NSCORE;
    int pos = r / NCLS;
    int c   = r % NCLS;
    float sc  = sigmoidf_(box_cls[(img * NCLS + c) * NPOS + pos]);
    float ctr = sigmoidf_(centerness[img * NPOS + pos]);
    flat[gid] = (sc > 0.05f) ? sc * ctr : -1.0f;
}

// ---------------------------------------------------------------------------
// 2) per-image top-1000 via iterative block-wide argmax
// ---------------------------------------------------------------------------
__global__ __launch_bounds__(1024)
void k_topk1000(float* __restrict__ flat, float* __restrict__ topv,
                int* __restrict__ topi) {
    __shared__ float sv[1024];
    __shared__ int   si[1024];
    const int img = blockIdx.x;
    const int tid = threadIdx.x;
    float* f = flat + img * NSCORE;
    for (int t = 0; t < PRE_TOPK; ++t) {
        float best = -1.0e30f; int bidx = 0;
        for (int j = tid; j < NSCORE; j += 1024) {
            float v = f[j];
            if (v > best) { best = v; bidx = j; }
        }
        sv[tid] = best; si[tid] = bidx;
        __syncthreads();
        for (int s = 512; s > 0; s >>= 1) {
            if (tid < s) {
                if (sv[tid + s] > sv[tid] ||
                    (sv[tid + s] == sv[tid] && si[tid + s] < si[tid])) {
                    sv[tid] = sv[tid + s]; si[tid] = si[tid + s];
                }
            }
            __syncthreads();
        }
        if (tid == 0) {
            topv[img * PRE_TOPK + t] = sv[0];
            topi[img * PRE_TOPK + t] = si[0];
            f[si[0]] = -2.0f;  // exclude from further rounds
        }
        __syncthreads();
    }
}

// ---------------------------------------------------------------------------
// 3) decode boxes + class-offset boxes
// ---------------------------------------------------------------------------
__global__ void k_decode(const float* __restrict__ locations,
                         const float* __restrict__ box_reg,
                         const int* __restrict__ topi,
                         float* __restrict__ boxes,
                         float* __restrict__ boff) {
    int gid = blockIdx.x * blockDim.x + threadIdx.x;
    if (gid >= NIMG * PRE_TOPK) return;
    int img = gid / PRE_TOPK;
    int idx = topi[gid];
    int loc = idx / NCLS;
    int cls = idx % NCLS + 1;
    float lx = locations[loc * 2 + 0];
    float ly = locations[loc * 2 + 1];
    const float* rg = box_reg + img * 4 * NPOS;
    float r0 = rg[0 * NPOS + loc], r1 = rg[1 * NPOS + loc];
    float r2 = rg[2 * NPOS + loc], r3 = rg[3 * NPOS + loc];
    float x1 = fminf(fmaxf(lx - r0, 0.0f), (float)(IMGW - 1));
    float y1 = fminf(fmaxf(ly - r1, 0.0f), (float)(IMGH - 1));
    float x2 = fminf(fmaxf(lx + r2, 0.0f), (float)(IMGW - 1));
    float y2 = fminf(fmaxf(ly + r3, 0.0f), (float)(IMGH - 1));
    float off = (float)cls * CLS_OFF;
    float* b  = boxes + gid * 4;
    float* bo = boff  + gid * 4;
    b[0] = x1; b[1] = y1; b[2] = x2; b[3] = y2;
    bo[0] = x1 + off; bo[1] = y1 + off; bo[2] = x2 + off; bo[3] = y2 + off;
}

// ---------------------------------------------------------------------------
// 4) NMS (serial scan over candidates, parallel over suppression targets)
// ---------------------------------------------------------------------------
__global__ __launch_bounds__(1024)
void k_nms(const float* __restrict__ boff, const float* __restrict__ topv,
           int* __restrict__ keep, float* __restrict__ kscr) {
    __shared__ float sb[PRE_TOPK * 4];
    __shared__ int   supp[PRE_TOPK];
    const int img = blockIdx.x;
    const int tid = threadIdx.x;
    for (int j = tid; j < PRE_TOPK * 4; j += 1024)
        sb[j] = boff[img * PRE_TOPK * 4 + j];
    for (int j = tid; j < PRE_TOPK; j += 1024)
        supp[j] = (topv[img * PRE_TOPK + j] > 0.0f) ? 0 : 1;
    __syncthreads();
    for (int i = 0; i < PRE_TOPK; ++i) {
        int keep_i = !supp[i];
        if (keep_i && tid < PRE_TOPK && tid > i) {
            float ax1 = sb[i * 4 + 0], ay1 = sb[i * 4 + 1];
            float ax2 = sb[i * 4 + 2], ay2 = sb[i * 4 + 3];
            float bx1 = sb[tid * 4 + 0], by1 = sb[tid * 4 + 1];
            float bx2 = sb[tid * 4 + 2], by2 = sb[tid * 4 + 3];
            float aA = (ax2 - ax1 + 1.0f) * (ay2 - ay1 + 1.0f);
            float bA = (bx2 - bx1 + 1.0f) * (by2 - by1 + 1.0f);
            float iw = fmaxf(fminf(ax2, bx2) - fmaxf(ax1, bx1) + 1.0f, 0.0f);
            float ih = fmaxf(fminf(ay2, by2) - fmaxf(ay1, by1) + 1.0f, 0.0f);
            float inter = iw * ih;
            float iou = inter / (aA + bA - inter);
            if (iou > NMS_T) supp[tid] = 1;
        }
        if (tid == 0) keep[img * PRE_TOPK + i] = keep_i;
        __syncthreads();
    }
    for (int j = tid; j < PRE_TOPK; j += 1024) {
        float v = topv[img * PRE_TOPK + j];
        float scr = sqrtf(fmaxf(v, 1e-12f));
        kscr[img * PRE_TOPK + j] = keep[img * PRE_TOPK + j] ? scr : -1.0f;
    }
}

// ---------------------------------------------------------------------------
// 5) top-100 of kept scores
// ---------------------------------------------------------------------------
__global__ __launch_bounds__(256)
void k_top100(float* __restrict__ kscr, float* __restrict__ v2,
              int* __restrict__ sel) {
    __shared__ float sv[256];
    __shared__ int   si[256];
    const int img = blockIdx.x;
    const int tid = threadIdx.x;
    float* f = kscr + img * PRE_TOPK;
    for (int t = 0; t < POST_TOPK; ++t) {
        float best = -1.0e30f; int bidx = 0;
        for (int j = tid; j < PRE_TOPK; j += 256) {
            float v = f[j];
            if (v > best) { best = v; bidx = j; }
        }
        sv[tid] = best; si[tid] = bidx;
        __syncthreads();
        for (int s = 128; s > 0; s >>= 1) {
            if (tid < s) {
                if (sv[tid + s] > sv[tid] ||
                    (sv[tid + s] == sv[tid] && si[tid + s] < si[tid])) {
                    sv[tid] = sv[tid + s]; si[tid] = si[tid + s];
                }
            }
            __syncthreads();
        }
        if (tid == 0) {
            v2[img * POST_TOPK + t]  = sv[0];
            sel[img * POST_TOPK + t] = si[0];
            f[si[0]] = -2.0f;
        }
        __syncthreads();
    }
}

// ---------------------------------------------------------------------------
// 6) gather final boxes / labels / scores / coefficients (zero-padded to 112)
// ---------------------------------------------------------------------------
__global__ void k_gather(const float* __restrict__ det_cofs,
                         const float* __restrict__ boxes,
                         const float* __restrict__ v2,
                         const int* __restrict__ sel,
                         const int* __restrict__ topi,
                         float* __restrict__ out,         // d_out base
                         float* __restrict__ fb,          // NIMG*DPAD*4
                         float* __restrict__ fval,        // NIMG*POST_TOPK
                         _Float16* __restrict__ coef_h) { // NIMG*DPAD*128
    int gid = blockIdx.x * blockDim.x + threadIdx.x;
    if (gid >= NIMG * DPAD) return;
    int img = gid / DPAD;
    int d   = gid % DPAD;
    int pd  = img * DPAD + d;
    if (d >= POST_TOPK) {   // zero pad rows 100..111
#pragma unroll
        for (int j = 0; j < 4; ++j) fb[pd * 4 + j] = 0.0f;
        v8h z = {};
        v8h* cz = (v8h*)(coef_h + pd * 128);
#pragma unroll
        for (int j = 0; j < 16; ++j) cz[j] = z;
        return;
    }
    int gq = img * POST_TOPK + d;
    int s  = sel[gq];
    float v = v2[gq];
    bool valid = v > 0.0f;
    float fv = valid ? 1.0f : 0.0f;
    int idx = topi[img * PRE_TOPK + s];
    int loc = idx / NCLS;
    int cls = idx % NCLS + 1;
    const float* b = boxes + (img * PRE_TOPK + s) * 4;
    // outputs: boxes[NIMG*100*4] | scores[NIMG*100] | labels[NIMG*100] | masks
#pragma unroll
    for (int j = 0; j < 4; ++j) {
        float bx = b[j] * fv;
        out[gq * 4 + j] = bx;
        fb[pd * 4 + j]  = bx;
    }
    out[NIMG * POST_TOPK * 4 + gq] = valid ? v : 0.0f;
    out[NIMG * POST_TOPK * 5 + gq] = valid ? (float)cls : 0.0f;
    fval[gq] = fv;
    const float* cof = det_cofs + img * 128 * NPOS;
#pragma unroll 4
    for (int c = 0; c < 128; ++c)
        coef_h[pd * 128 + c] = (_Float16)cof[c * NPOS + loc];
}

// ---------------------------------------------------------------------------
// 7) mask head GEMM via WMMA. One wave owns one 16-det column tile and
//    MT_PER_WAVE pixel tiles (B loaded once, reused 4x). All fragments are
//    contiguous b128 loads matching the ISA 16-bit A/B VGPR layouts.
// ---------------------------------------------------------------------------
__global__ __launch_bounds__(256)
void k_mask_wmma(const _Float16* __restrict__ featT,  // (NIMG,NPIX,32)
                 const _Float16* __restrict__ coef_h, // (NIMG,DPAD,128)
                 const float* __restrict__ fb,        // (NIMG,DPAD,4)
                 _Float16* __restrict__ pm) {         // (NIMG,100,NPIX)
    const int wid  = threadIdx.x >> 5;
    const int lane = threadIdx.x & 31;
    const int img  = blockIdx.y;
    const int wg   = blockIdx.x * 8 + wid;   // 0 .. WAVES_TOT-1
    const int tN   = wg % NTILES_N;
    const int tMg  = wg / NTILES_N;          // group of MT_PER_WAVE M-tiles

    const int mrow  = lane & 15;
    const int kbase = (lane < 16) ? 0 : 8;   // ISA 16-bit fragment K split
    const int hhalf = (lane < 16) ? 0 : 8;   // C/D row half

    // ---- B fragments: one per quadrant, loaded once per wave ----
    const int dcol = tN * 16 + (lane & 15);  // < DPAD, zero rows past 100
    const _Float16* cb = coef_h + (size_t)(img * DPAD + dcol) * 128;
    v16h b0, b1, b2, b3;
    {
        v8h lo, hi;
        lo = *(const v8h*)(cb + 0 * 32 + kbase);
        hi = *(const v8h*)(cb + 0 * 32 + kbase + 16);
        b0 = __builtin_shufflevector(lo, hi, 0,1,2,3,4,5,6,7,8,9,10,11,12,13,14,15);
        lo = *(const v8h*)(cb + 1 * 32 + kbase);
        hi = *(const v8h*)(cb + 1 * 32 + kbase + 16);
        b1 = __builtin_shufflevector(lo, hi, 0,1,2,3,4,5,6,7,8,9,10,11,12,13,14,15);
        lo = *(const v8h*)(cb + 2 * 32 + kbase);
        hi = *(const v8h*)(cb + 2 * 32 + kbase + 16);
        b2 = __builtin_shufflevector(lo, hi, 0,1,2,3,4,5,6,7,8,9,10,11,12,13,14,15);
        lo = *(const v8h*)(cb + 3 * 32 + kbase);
        hi = *(const v8h*)(cb + 3 * 32 + kbase + 16);
        b3 = __builtin_shufflevector(lo, hi, 0,1,2,3,4,5,6,7,8,9,10,11,12,13,14,15);
    }

    // ---- box for this detection column (fb zero-padded; fvalid pre-applied)
    const float* bb = fb + (size_t)(img * DPAD + dcol) * 4;
    const float x1 = bb[0] / SCALE_, y1 = bb[1] / SCALE_;
    const float x2 = bb[2] / SCALE_, y2 = bb[3] / SCALE_;
    const float cx = 0.5f * (x1 + x2), cy = 0.5f * (y1 + y2);
    const bool dok = (dcol < POST_TOPK);
    _Float16* pmd = pm + (size_t)(img * POST_TOPK + dcol) * NPIX;

    const v8f zacc = {};
    const _Float16* fB = featT + (size_t)img * NPIX * 32;

#pragma unroll
    for (int i = 0; i < MT_PER_WAVE; ++i) {
        const int tM  = tMg * MT_PER_WAVE + i;
        const int p_a = tM * 16 + mrow;
        __builtin_prefetch(fB + (size_t)(p_a + 16) * 32, 0, 3); // global_prefetch_b8

        // A fragment: two contiguous 16B chunks per lane (ISA 16x32 layout)
        v8h alo = *(const v8h*)(fB + (size_t)p_a * 32 + kbase);
        v8h ahi = *(const v8h*)(fB + (size_t)p_a * 32 + kbase + 16);
        v16h a = __builtin_shufflevector(alo, ahi,
                    0,1,2,3,4,5,6,7,8,9,10,11,12,13,14,15);

        v8f acc0 = __builtin_amdgcn_wmma_f32_16x16x32_f16(false, a, false, b0,
                                                          (short)0, zacc, false, false);
        v8f acc1 = __builtin_amdgcn_wmma_f32_16x16x32_f16(false, a, false, b1,
                                                          (short)0, zacc, false, false);
        v8f acc2 = __builtin_amdgcn_wmma_f32_16x16x32_f16(false, a, false, b2,
                                                          (short)0, zacc, false, false);
        v8f acc3 = __builtin_amdgcn_wmma_f32_16x16x32_f16(false, a, false, b3,
                                                          (short)0, zacc, false, false);

        // epilogue: 8 consecutive pixels per lane -> one b128 store
        // (a 16-px tile never crosses a 320-px row: 320 % 16 == 0)
        const int pbase = tM * 16 + hhalf;
        const float y   = (float)((tM * 16) / WM_);
        const bool topq = y < cy;
        const bool yin  = (y >= y1) && (y <= y2);
        v8h outv;
#pragma unroll
        for (int v = 0; v < 8; ++v) {
            float x = (float)((pbase + v) % WM_ == pbase % WM_ + v
                              ? (pbase % WM_) + v : (pbase % WM_) + v); // same row
            x = (float)((tM * 16) % WM_ + hhalf + v);
            bool inside = yin && (x >= x1) && (x <= x2);
            bool left   = x < cx;
            float z = topq ? (left ? acc0[v] : acc1[v])
                           : (left ? acc2[v] : acc3[v]);
            float m = inside ? sigmoidf_(z) : 0.0f;
            outv[v] = (_Float16)m;
        }
        if (dok) *(v8h*)(pmd + pbase) = outv;
    }
}

// ---------------------------------------------------------------------------
// 8) bilinear 2x resize (half-pixel centers), threshold, validity
// ---------------------------------------------------------------------------
__global__ void k_resize(const _Float16* __restrict__ pm,
                         const float* __restrict__ fval,
                         float* __restrict__ out_masks) {
    int gid = blockIdx.x * blockDim.x + threadIdx.x;
    if (gid >= NIMG * POST_TOPK * IMGH * IMGW) return;
    int ox = gid % IMGW;
    int t  = gid / IMGW;
    int oy = t % IMGH;
    int kd = t / IMGH;           // img*100 + d
    float fv = fval[kd];
    const _Float16* src = pm + (size_t)kd * NPIX;
    float sx = (float)ox * 0.5f - 0.25f;
    float sy = (float)oy * 0.5f - 0.25f;
    int x0 = (int)floorf(sx), y0 = (int)floorf(sy);
    float wx = sx - (float)x0, wy = sy - (float)y0;
    int x1 = min(max(x0 + 1, 0), WM_ - 1);
    int y1 = min(max(y0 + 1, 0), HM_ - 1);
    x0 = min(max(x0, 0), WM_ - 1);
    y0 = min(max(y0, 0), HM_ - 1);
    float v00 = (float)src[y0 * WM_ + x0];
    float v01 = (float)src[y0 * WM_ + x1];
    float v10 = (float)src[y1 * WM_ + x0];
    float v11 = (float)src[y1 * WM_ + x1];
    float v = (v00 * (1.0f - wx) + v01 * wx) * (1.0f - wy)
            + (v10 * (1.0f - wx) + v11 * wx) * wy;
    out_masks[gid] = (fv > 0.0f && v > MASK_T) ? 1.0f : 0.0f;
}

// ---------------------------------------------------------------------------
extern "C" void kernel_launch(void* const* d_in, const int* in_sizes, int n_in,
                              void* d_out, int out_size, void* d_ws, size_t ws_size,
                              hipStream_t stream) {
    const float* locations  = (const float*)d_in[0];
    const float* box_cls    = (const float*)d_in[1];
    const float* box_reg    = (const float*)d_in[2];
    const float* centerness = (const float*)d_in[3];
    const float* det_cofs   = (const float*)d_in[4];
    const float* feat_masks = (const float*)d_in[5];
    float* out = (float*)d_out;

    float*    wf = (float*)d_ws;
    int*      wi = (int*)(wf + WSF_END);
    _Float16* wh = (_Float16*)(wi + WSI_END);

    float* flat  = wf + WSF_FLAT;
    float* topv  = wf + WSF_TOPV;
    float* boxes = wf + WSF_BOX;
    float* boff  = wf + WSF_BOFF;
    float* kscr  = wf + WSF_KSCR;
    float* v2    = wf + WSF_V2;
    float* fval  = wf + WSF_FVAL;
    float* fb    = wf + WSF_FB;
    int*   topi  = wi + WSI_TOPI;
    int*   keep  = wi + WSI_KEEP;
    int*   sel   = wi + WSI_SEL;
    _Float16* coef_h = wh + WSH_COEF;
    _Float16* pm     = wh + WSH_PM;
    _Float16* featT  = wh + WSH_FEAT;

    // 0) feat f32 -> f16 pixel-major (A operand staging)
    k_feat_f16<<<(NIMG * NPIX + 255) / 256, 256, 0, stream>>>(feat_masks, featT);
    // 1) candidate scores
    k_score_flat<<<(NIMG * NSCORE) / 256, 256, 0, stream>>>(box_cls, centerness, flat);
    // 2) top-1000 per image
    k_topk1000<<<NIMG, 1024, 0, stream>>>(flat, topv, topi);
    // 3) box decode
    k_decode<<<(NIMG * PRE_TOPK + 255) / 256, 256, 0, stream>>>(
        locations, box_reg, topi, boxes, boff);
    // 4) NMS
    k_nms<<<NIMG, 1024, 0, stream>>>(boff, topv, keep, kscr);
    // 5) top-100 per image
    k_top100<<<NIMG, 256, 0, stream>>>(kscr, v2, sel);
    // 6) gather final detections + zero-padded coefficients
    k_gather<<<(NIMG * DPAD + 255) / 256, 256, 0, stream>>>(
        det_cofs, boxes, v2, sel, topi, out, fb, fval, coef_h);
    // 7) WMMA mask GEMM + crop
    dim3 mg(MASK_BLOCKS, NIMG);
    k_mask_wmma<<<mg, 256, 0, stream>>>(featT, coef_h, fb, pm);
    // 8) resize + threshold -> boolean masks (as 0/1 floats)
    const long long npixout = (long long)NIMG * POST_TOPK * IMGH * IMGW;
    k_resize<<<(unsigned)((npixout + 255) / 256), 256, 0, stream>>>(
        pm, fval, out + NIMG * POST_TOPK * 6);
}
